// Attention_44040594653486
// MI455X (gfx1250) — matrix operand order repaired and checked by reading, not compiled, not run on hardware
//
#include <hip/hip_runtime.h>

#define BATCH 4
#define CH    32
#define SPV   (64*64*64)          // voxels per channel = 262144
#define NBLK  (SPV/32)            // 8192 blocks of 32 voxels per batch
#define WAVES 8
#define NWG1  64
#define NWG2  64
#define PITCH 40                  // bf16 elems per LDS row (80 B, keeps 16B alignment)

typedef __attribute__((ext_vector_type(16))) __bf16 v16bf;
typedef __attribute__((ext_vector_type(8)))  __bf16 v8bf;
typedef __attribute__((ext_vector_type(8)))  float  v8f;

static __device__ __forceinline__ v8f wmma_bf16(v16bf a, v16bf b, v8f c) {
  // D(f32 16x16) = A(bf16 16x32) * B(bf16 32x16) + C(f32)
  return __builtin_amdgcn_wmma_f32_16x16x32_bf16(false, a, false, b, (short)0, c, false, false);
}

static __device__ __forceinline__ v8f vzero8() {
  v8f z;
#pragma unroll
  for (int r = 0; r < 8; ++r) z[r] = 0.0f;
  return z;
}

// A-fragment (16x32) of a 32x32 row-major f32 weight matrix, row-tile mt.
// Lane L: M = mt*16 + (L&15); slot e -> K = 16h+e.
static __device__ __forceinline__ v16bf load_w_afrag(const float* __restrict__ W, int mt, int lane) {
  const int row  = mt * 16 + (lane & 15);
  const int half = lane >> 4;
  v16bf f;
#pragma unroll
  for (int e = 0; e < 8; ++e) {
    f[e]     = (__bf16)W[row * 32 + 16 * half + e];
    f[8 + e] = (__bf16)W[row * 32 + 16 * half + 8 + e];
  }
  return f;
}

// C-fragment broadcasting bias over the N dimension: c[r] = bias[mt*16 + 8h + r]
static __device__ __forceinline__ v8f load_bias_cfrag(const float* __restrict__ bvec, int mt, int lane) {
  const int half = lane >> 4;
  v8f c;
#pragma unroll
  for (int r = 0; r < 8; ++r) c[r] = bvec[mt * 16 + 8 * half + r];
  return c;
}

// B-fragment (K=cin 32 x N=sp 16) of x, converted f32->bf16 on the fly.
// Lane L: N = col0 + (L&15); slot e -> K = 16h + e.
// Each generated global_load_b32 fetches two coalesced 64B segments.
static __device__ __forceinline__ v16bf load_x_bfrag(const float* __restrict__ xb, int col0, int lane) {
  const int half = lane >> 4;
  const int col  = col0 + (lane & 15);
  v16bf f;
#pragma unroll
  for (int e = 0; e < 16; ++e)
    f[e] = (__bf16)xb[(size_t)(16 * half + e) * SPV + col];
  return f;
}

// Store a D-tile (mt=row-tile, nt=col-tile of a 32x32 result) TRANSPOSED into LDS
// T[sp][ch] (pitch PITCH bf16). Per lane: one packed 16-byte ds_store_b128.
static __device__ __forceinline__ void store_tile_T(__bf16* T, int mt, int nt, v8f d, int lane) {
  const int half = lane >> 4;
  const int sp   = nt * 16 + (lane & 15);
  const int c0   = mt * 16 + 8 * half;
  v8bf p;
#pragma unroll
  for (int r = 0; r < 8; ++r) p[r] = (__bf16)d[r];
  *(v8bf*)(T + sp * PITCH + c0) = p;
}

// A-fragment from an LDS matrix T[row][k] (row-major, pitch PITCH): 2x ds_load_b128.
static __device__ __forceinline__ v16bf load_afrag_lds(const __bf16* T, int mt, int lane) {
  const int row  = mt * 16 + (lane & 15);
  const int half = lane >> 4;
  v8bf lo = *(const v8bf*)(T + row * PITCH + 16 * half);        // K = 16h .. 16h+7
  v8bf hi = *(const v8bf*)(T + row * PITCH + 16 * half + 8);    // K = 16h+8 ..
  v16bf f;
#pragma unroll
  for (int e = 0; e < 8; ++e) { f[e] = lo[e]; f[8 + e] = hi[e]; }
  return f;
}

// B-fragment from LDS T[n][k] (row = N index): K = 16h + e -> 2x ds_load_b128.
static __device__ __forceinline__ v16bf load_bfrag_lds(const __bf16* T, int nt, int lane) {
  const int row  = nt * 16 + (lane & 15);
  const int half = lane >> 4;
  v8bf lo = *(const v8bf*)(T + row * PITCH + 16 * half);
  v8bf hi = *(const v8bf*)(T + row * PITCH + 16 * half + 8);
  v16bf f;
#pragma unroll
  for (int e = 0; e < 8; ++e) { f[e] = lo[e]; f[8 + e] = hi[e]; }
  return f;
}

// ---------------- Pass 1: s[b] += (Wq X + bq)^T (Wk X + bk) per 32-voxel block --------------
__global__ void __launch_bounds__(256)
pass1_qk(const float* __restrict__ x,
         const float* __restrict__ Wq, const float* __restrict__ bq,
         const float* __restrict__ Wk, const float* __restrict__ bk,
         float* __restrict__ sOut) {
  __shared__ alignas(16) char smem[WAVES * 2 * 32 * PITCH * 2];  // 40 KiB bf16 staging / f32 reduce
  const int lane = threadIdx.x & 31;
  const int wave = threadIdx.x >> 5;
  const int b    = blockIdx.x / NWG1;
  const int wg   = blockIdx.x % NWG1;
  const float* xb = x + (size_t)b * CH * SPV;

  __bf16* Qt = (__bf16*)smem + (size_t)wave * 2 * 32 * PITCH;  // wave-private Q^T staging
  __bf16* Kt = Qt + 32 * PITCH;

  v16bf wqa[2], wka[2];
  v8f   cq[2],  ck[2];
#pragma unroll
  for (int mt = 0; mt < 2; ++mt) {
    wqa[mt] = load_w_afrag(Wq, mt, lane);
    wka[mt] = load_w_afrag(Wk, mt, lane);
    cq[mt]  = load_bias_cfrag(bq, mt, lane);
    ck[mt]  = load_bias_cfrag(bk, mt, lane);
  }

  v8f acc[2][2];
#pragma unroll
  for (int i = 0; i < 2; ++i)
#pragma unroll
    for (int j = 0; j < 2; ++j) acc[i][j] = vzero8();

  const int gw     = wg * WAVES + wave;
  const int stride = NWG1 * WAVES;
  for (int blk = gw; blk < NBLK; blk += stride) {
    const int sp0 = blk * 32;
    if (blk + stride < NBLK)  // pull next block's line for this lane's channel row
      __builtin_prefetch(xb + (size_t)lane * SPV + (size_t)(blk + stride) * 32, 0, 1);

    v16bf xf0 = load_x_bfrag(xb, sp0, lane);
    v16bf xf1 = load_x_bfrag(xb, sp0 + 16, lane);

    v8f q[2][2], k[2][2];
    q[0][0] = wmma_bf16(wqa[0], xf0, cq[0]);
    q[0][1] = wmma_bf16(wqa[0], xf1, cq[0]);
    q[1][0] = wmma_bf16(wqa[1], xf0, cq[1]);
    q[1][1] = wmma_bf16(wqa[1], xf1, cq[1]);
    k[0][0] = wmma_bf16(wka[0], xf0, ck[0]);
    k[0][1] = wmma_bf16(wka[0], xf1, ck[0]);
    k[1][0] = wmma_bf16(wka[1], xf0, ck[1]);
    k[1][1] = wmma_bf16(wka[1], xf1, ck[1]);

#pragma unroll
    for (int mt = 0; mt < 2; ++mt)
#pragma unroll
      for (int nt = 0; nt < 2; ++nt) {
        store_tile_T(Qt, mt, nt, q[mt][nt], lane);
        store_tile_T(Kt, mt, nt, k[mt][nt], lane);
      }
    __builtin_amdgcn_wave_barrier();   // wave-private LDS: ordering only

    v16bf aq0 = load_afrag_lds(Qt, 0, lane);  // A = Q^T (M = sp-in-block, K = ch)
    v16bf aq1 = load_afrag_lds(Qt, 1, lane);
    v16bf bk0 = load_bfrag_lds(Kt, 0, lane);  // B = K   (K = ch, N = sp-in-block)
    v16bf bk1 = load_bfrag_lds(Kt, 1, lane);
    acc[0][0] = wmma_bf16(aq0, bk0, acc[0][0]);
    acc[0][1] = wmma_bf16(aq0, bk1, acc[0][1]);
    acc[1][0] = wmma_bf16(aq1, bk0, acc[1][0]);
    acc[1][1] = wmma_bf16(aq1, bk1, acc[1][1]);
    __builtin_amdgcn_wave_barrier();
  }

  // cross-wave reduce in LDS, then one atomic per (workgroup, element)
  __syncthreads();
  float* red = (float*)smem;  // 8 waves * 1024 f32 = 32 KiB
  const int half = lane >> 4;
#pragma unroll
  for (int it = 0; it < 2; ++it)
#pragma unroll
    for (int jt = 0; jt < 2; ++jt)
#pragma unroll
      for (int r = 0; r < 8; ++r) {
        const int i = it * 16 + 8 * half + r;
        const int j = jt * 16 + (lane & 15);
        red[wave * 1024 + i * 32 + j] = acc[it][jt][r];
      }
  __syncthreads();
  float* sb = sOut + b * 1024;
  for (int e = threadIdx.x; e < 1024; e += 256) {
    float sum = 0.0f;
#pragma unroll
    for (int w = 0; w < WAVES; ++w) sum += red[w * 1024 + e];
    atomicAdd(&sb[e], sum);
  }
}

// ---------------- softmax over last dim of s (4 x 32 x 32) -------------------
__global__ void __launch_bounds__(32)
softmax32(const float* __restrict__ s, float* __restrict__ a) {
  const int b = blockIdx.x;
  const int i = threadIdx.x;  // one lane per row
  const float* row = s + b * 1024 + i * 32;
  float m = row[0];
#pragma unroll
  for (int j = 1; j < 32; ++j) m = fmaxf(m, row[j]);
  float e[32], sum = 0.0f;
#pragma unroll
  for (int j = 0; j < 32; ++j) { e[j] = __expf(row[j] - m); sum += e[j]; }
  const float inv = 1.0f / sum;
  float* ar = a + b * 1024 + i * 32;
#pragma unroll
  for (int j = 0; j < 32; ++j) ar[j] = e[j] * inv;
}

// ---------------- Pass 2: out = x + gamma * ((Wv X + bv) * a) ----------------
__global__ void __launch_bounds__(256)
pass2_out(const float* __restrict__ x,
          const float* __restrict__ Wv, const float* __restrict__ bv,
          const float* __restrict__ aMat, const float* __restrict__ gammaPtr,
          float* __restrict__ out) {
  __shared__ alignas(16) __bf16 Vs[WAVES][32 * PITCH];  // 20 KiB, wave-private V staging
  const int lane = threadIdx.x & 31;
  const int wave = threadIdx.x >> 5;
  const int b    = blockIdx.x / NWG2;
  const int wg   = blockIdx.x % NWG2;
  const float* xb = x   + (size_t)b * CH * SPV;
  float*       ob = out + (size_t)b * CH * SPV;
  const float  g  = gammaPtr[0];

  v16bf wva[2];
  v8f   cv[2];
#pragma unroll
  for (int mt = 0; mt < 2; ++mt) {
    wva[mt] = load_w_afrag(Wv, mt, lane);
    cv[mt]  = load_bias_cfrag(bv, mt, lane);
  }
  // a (32x32 f32, row-major) as B-fragments: B[k=i][n=j]
  const float* ab = aMat + b * 1024;
  const int half = lane >> 4;
  v16bf af[2];
#pragma unroll
  for (int jt = 0; jt < 2; ++jt) {
    const int n = jt * 16 + (lane & 15);
#pragma unroll
    for (int e = 0; e < 16; ++e) af[jt][e] = (__bf16)ab[(16 * half + e) * 32 + n];
  }

  __bf16* V = &Vs[wave][0];
  const int gw     = wg * WAVES + wave;
  const int stride = NWG2 * WAVES;
  for (int blk = gw; blk < NBLK; blk += stride) {
    const int sp0 = blk * 32;
    if (blk + stride < NBLK)
      __builtin_prefetch(xb + (size_t)lane * SPV + (size_t)(blk + stride) * 32, 0, 1);

    v16bf xf0 = load_x_bfrag(xb, sp0, lane);
    v16bf xf1 = load_x_bfrag(xb, sp0 + 16, lane);

    v8f v[2][2];
    v[0][0] = wmma_bf16(wva[0], xf0, cv[0]);
    v[0][1] = wmma_bf16(wva[0], xf1, cv[0]);
    v[1][0] = wmma_bf16(wva[1], xf0, cv[1]);
    v[1][1] = wmma_bf16(wva[1], xf1, cv[1]);

    // stage V row-major [ch][i] so it can be re-read as the next GEMM's A-matrix
#pragma unroll
    for (int mt = 0; mt < 2; ++mt)
#pragma unroll
      for (int nt = 0; nt < 2; ++nt)
#pragma unroll
        for (int r = 0; r < 8; ++r)
          V[(mt * 16 + 8 * half + r) * PITCH + nt * 16 + (lane & 15)] = (__bf16)v[mt][nt][r];
    __builtin_amdgcn_wave_barrier();

    v16bf av0 = load_afrag_lds(V, 0, lane);
    v16bf av1 = load_afrag_lds(V, 1, lane);
    const v8f z = vzero8();
    v8f o[2][2];
    o[0][0] = wmma_bf16(av0, af[0], z);
    o[0][1] = wmma_bf16(av0, af[1], z);
    o[1][0] = wmma_bf16(av1, af[0], z);
    o[1][1] = wmma_bf16(av1, af[1], z);

    // fused residual: out = x + gamma * scores
#pragma unroll
    for (int mt = 0; mt < 2; ++mt)
#pragma unroll
      for (int nt = 0; nt < 2; ++nt)
#pragma unroll
        for (int r = 0; r < 8; ++r) {
          const int ch = mt * 16 + 8 * half + r;
          const size_t idx = (size_t)ch * SPV + sp0 + nt * 16 + (lane & 15);
          ob[idx] = xb[idx] + g * o[mt][nt][r];
        }
    __builtin_amdgcn_wave_barrier();
  }
}

extern "C" void kernel_launch(void* const* d_in, const int* in_sizes, int n_in,
                              void* d_out, int out_size, void* d_ws, size_t ws_size,
                              hipStream_t stream) {
  const float* x  = (const float*)d_in[0];
  const float* Wq = (const float*)d_in[1];
  const float* bq = (const float*)d_in[2];
  const float* Wk = (const float*)d_in[3];
  const float* bk = (const float*)d_in[4];
  const float* Wv = (const float*)d_in[5];
  const float* bv = (const float*)d_in[6];
  const float* gm = (const float*)d_in[7];
  float* out  = (float*)d_out;
  float* sbuf = (float*)d_ws;             // [4][32][32] gram accumulators
  float* abuf = sbuf + BATCH * 1024;      // [4][32][32] softmax(a)

  hipMemsetAsync(sbuf, 0, BATCH * 1024 * sizeof(float), stream);
  hipLaunchKernelGGL(pass1_qk, dim3(BATCH * NWG1), dim3(256), 0, stream,
                     x, Wq, bq, Wk, bk, sbuf);
  hipLaunchKernelGGL(softmax32, dim3(BATCH), dim3(32), 0, stream, sbuf, abuf);
  hipLaunchKernelGGL(pass2_out, dim3(BATCH * NWG2), dim3(256), 0, stream,
                     x, Wv, bv, abuf, gm, out);
}
